// TimeScaledAttention_12395275616611
// MI455X (gfx1250) — matrix operand order repaired and checked by reading, not compiled, not run on hardware
//
#include <hip/hip_runtime.h>
#include <math.h>

// ---------------------------------------------------------------------------
// TimeScaledAttention for MI455X (gfx1250, wave32, WMMA)
//   B=8, N=4096, DIM=128, HID=64, fp32 in/out, bf16 WMMA internally.
//   ts(q) = sigmoid(MLP(time_gap)) folded into logit scale: ts[q]/sqrt(D).
//   V is stored TRANSPOSED [B][DIM][N] by the projection kernel so the
//   attention kernel can stage both K and V^T tiles with contiguous
//   GLOBAL_LOAD_ASYNC_TO_LDS_B128 (double-buffered, ASYNCcnt-tracked).
// ---------------------------------------------------------------------------

#define BB   8
#define NN   4096
#define DIM  128
#define HID  64

typedef __attribute__((ext_vector_type(16))) __bf16 v16bf;
typedef __attribute__((ext_vector_type(8)))  __bf16 v8bf;
typedef __attribute__((ext_vector_type(8)))  float  v8f;
typedef __attribute__((ext_vector_type(4)))  int    v4i;

union BF16x16 { v16bf v; v8bf h[2]; };

#if __has_builtin(__builtin_amdgcn_global_load_async_to_lds_b128)
#define HAVE_ASYNC 1
typedef __attribute__((address_space(1))) v4i* as1_v4i_ptr;
typedef __attribute__((address_space(3))) v4i* as3_v4i_ptr;
#else
#define HAVE_ASYNC 0
#endif

// 16-byte global->LDS copy: async (ASYNCcnt) when available, else sync.
__device__ __forceinline__ void cp16_g2l(const __bf16* g, __bf16* l) {
#if HAVE_ASYNC
  __builtin_amdgcn_global_load_async_to_lds_b128(
      (as1_v4i_ptr)g, (as3_v4i_ptr)l, /*offset=*/0, /*cpol=*/0);
#else
  *(v8bf*)l = *(const v8bf*)g;
#endif
}

__device__ __forceinline__ void wait_async_all() {
#if HAVE_ASYNC
  asm volatile("s_wait_asynccnt 0" ::: "memory");
#endif
}

// ---------------------------------------------------------------------------
// Kernel 1: time-gap MLP -> ts[B*N]  (tiny; VALU)
// ---------------------------------------------------------------------------
__global__ void __launch_bounds__(256)
ts_kernel(const float* __restrict__ tg,
          const float* __restrict__ W1, const float* __restrict__ b1,
          const float* __restrict__ W2, const float* __restrict__ b2,
          const float* __restrict__ W3, const float* __restrict__ b3,
          float* __restrict__ ts) {
  __shared__ float W2s[HID * HID];
  __shared__ float w1s[HID], b1s[HID], b2s[HID], w3s[HID];
  int t = threadIdx.x;
  for (int i = t; i < HID * HID; i += blockDim.x) W2s[i] = W2[i];
  if (t < HID) { w1s[t] = W1[t]; b1s[t] = b1[t]; b2s[t] = b2[t]; w3s[t] = W3[t]; }
  __syncthreads();

  int idx = blockIdx.x * blockDim.x + t;
  float x = tg[idx];
  float h1[HID];
#pragma unroll
  for (int j = 0; j < HID; ++j) h1[j] = fmaxf(fmaf(x, w1s[j], b1s[j]), 0.f);
  float acc = b3[0];
  for (int j = 0; j < HID; ++j) {        // outer rolled, inner unrolled
    float h2 = b2s[j];
#pragma unroll
    for (int i2 = 0; i2 < HID; ++i2) h2 = fmaf(h1[i2], W2s[i2 * HID + j], h2);
    h2 = fmaxf(h2, 0.f);
    acc = fmaf(h2, w3s[j], acc);
  }
  ts[idx] = 1.f / (1.f + __expf(-acc));
}

// ---------------------------------------------------------------------------
// Kernel 2: Q/K/V projections, bf16 out.  Block = 128 thr (4 waves), 64 rows.
//   Q, K row-major [B*N][DIM];  V transposed [B][DIM][N].
// ---------------------------------------------------------------------------
__global__ void __launch_bounds__(128)
proj_kernel(const float* __restrict__ x,
            const float* __restrict__ Wq, const float* __restrict__ bq,
            const float* __restrict__ Wk, const float* __restrict__ bk,
            const float* __restrict__ Wv, const float* __restrict__ bv,
            __bf16* __restrict__ Qo, __bf16* __restrict__ Ko, __bf16* __restrict__ Vo) {
  __shared__ __bf16 xt[64][DIM];      // 16 KB, row-major x tile (bf16)
  __shared__ __bf16 Wt[DIM][DIM];     // 32 KB, transposed weights (bf16)

  int t = threadIdx.x, lane = t & 31, wave = t >> 5;
  int lcol = lane & 15, lhalf = lane >> 4;
  long r0 = (long)blockIdx.x * 64;
  long bB = r0 / NN;                  // batch index (64 | NN, no straddle)

  { // stage x tile (f32 -> bf16)
    int row = t >> 1, half = t & 1;
    const float* src = x + (r0 + row) * DIM + half * 64;
#pragma unroll 8
    for (int j = 0; j < 64; ++j) xt[row][half * 64 + j] = (__bf16)src[j];
  }
  { // stage Wq transposed
    for (int din = 0; din < DIM; ++din) Wt[t][din] = (__bf16)Wq[din * DIM + t];
  }
  __syncthreads();

  // A fragments (16 rows of x), ISA A-layout: lane=M, K 0-7/16-23 | 8-15/24-31
  BF16x16 af[4];
#pragma unroll
  for (int ch = 0; ch < 4; ++ch) {
    const __bf16* p = &xt[wave * 16 + lcol][ch * 32 + lhalf * 8];
    af[ch].h[0] = *(const v8bf*)p;
    af[ch].h[1] = *(const v8bf*)(p + 16);
  }

  const float* Ws[3] = {Wq, Wk, Wv};
  const float* bs[3] = {bq, bk, bv};

  for (int m = 0; m < 3; ++m) {
#pragma unroll
    for (int dt = 0; dt < 8; ++dt) {
      v8f c = {};
#pragma unroll
      for (int ch = 0; ch < 4; ++ch) {
        BF16x16 bf;
        const __bf16* p = &Wt[dt * 16 + lcol][ch * 32 + lhalf * 8];
        bf.h[0] = *(const v8bf*)p;
        bf.h[1] = *(const v8bf*)(p + 16);
        c = __builtin_amdgcn_wmma_f32_16x16x32_bf16(false, af[ch].v, false, bf.v,
                                                    (short)0, c, false, false);
      }
      int   dcol = dt * 16 + lcol;
      float bias = bs[m][dcol];
      long  rowb = r0 + wave * 16 + lhalf * 8;
      if (m == 0) {
#pragma unroll
        for (int r = 0; r < 8; ++r) Qo[(rowb + r) * DIM + dcol] = (__bf16)(c[r] + bias);
      } else if (m == 1) {
#pragma unroll
        for (int r = 0; r < 8; ++r) Ko[(rowb + r) * DIM + dcol] = (__bf16)(c[r] + bias);
      } else {  // V: store transposed [B][DIM][N]
        long n0 = rowb - bB * NN;
        __bf16* vp = Vo + (bB * DIM + dcol) * NN + n0;
#pragma unroll
        for (int r = 0; r < 8; ++r) vp[r] = (__bf16)(c[r] + bias);
      }
    }
    if (m < 2) {
      __syncthreads();  // everyone done reading Wt
      for (int din = 0; din < DIM; ++din) Wt[t][din] = (__bf16)Ws[m + 1][din * DIM + t];
      __syncthreads();
    }
  }
}

// ---------------------------------------------------------------------------
// Kernel 3: flash attention.  Block = 128 thr (4 waves x 16 queries = 64 q).
// Double-buffered async LDS staging of K (row-major) and V^T (d-major) tiles.
// ---------------------------------------------------------------------------
__global__ void __launch_bounds__(128)
attn_kernel(const __bf16* __restrict__ Q, const __bf16* __restrict__ K,
            const __bf16* __restrict__ Vtg,   // transposed V: [B][DIM][N]
            const float* __restrict__ ts, float* __restrict__ out) {
  __shared__ __bf16 Kt[2][32][DIM];   // 2 x 8 KB  keys row-major
  __shared__ __bf16 Vt[2][DIM][32];   // 2 x 8 KB  V^T tile (d-major)
  __shared__ __bf16 Pt[4][16][32];    // 4 KB  per-wave P staging (C->A relayout)

  int t = threadIdx.x, lane = t & 31, wave = t >> 5;
  int lcol = lane & 15, lhalf = lane >> 4;

  int nqt = NN / 64;
  int b   = blockIdx.x / nqt;
  int qt  = blockIdx.x % nqt;
  long q0 = (long)b * NN + qt * 64 + wave * 16;  // flat query row base (this wave)
  long kb = (long)b * NN;                        // flat key row base
  const __bf16* Vg = Vtg + (long)b * DIM * NN;   // this batch's V^T

  // stage one 32-key tile (K rows + V^T rows) into buffer `bufi`
  auto stage = [&](int bufi, int kk2) {
    int krow = t >> 2, seg = t & 3;              // 32 key rows x 4 segments
    const __bf16* ks = K + (kb + kk2 + krow) * DIM + seg * 32;
    __bf16* kd = &Kt[bufi][krow][seg * 32];
#pragma unroll
    for (int j = 0; j < 4; ++j) cp16_g2l(ks + j * 8, kd + j * 8);
    const __bf16* vs = Vg + (long)t * NN + kk2;  // d = t, 32 contiguous keys
    __bf16* vd = &Vt[bufi][t][0];
#pragma unroll
    for (int j = 0; j < 4; ++j) cp16_g2l(vs + j * 8, vd + j * 8);
  };

  // Q fragments: 16 x 128 bf16 -> 4 A-frags
  BF16x16 qf[4];
#pragma unroll
  for (int ch = 0; ch < 4; ++ch) {
    const __bf16* p = Q + (q0 + lcol) * DIM + ch * 32 + lhalf * 8;
    qf[ch].h[0] = *(const v8bf*)p;
    qf[ch].h[1] = *(const v8bf*)(p + 16);
  }
  // per-row logit scale ts[q]/sqrt(D)
  float tsv[8];
#pragma unroll
  for (int r = 0; r < 8; ++r)
    tsv[r] = ts[q0 + lhalf * 8 + r] * 0.08838834764831845f;

  float mrow[8], lrow[8];
  v8f   acc[8] = {};
#pragma unroll
  for (int r = 0; r < 8; ++r) { mrow[r] = -1e30f; lrow[r] = 0.f; }

  stage(0, 0);  // prologue: issue tile 0

  for (int kk = 0; kk < NN; kk += 32) {
    int buf = (kk >> 5) & 1;
    wait_async_all();     // this wave's portion of tile `buf` is in LDS
    __syncthreads();      // all waves' portions are in LDS / prev reads done
    if (kk + 32 < NN) stage(buf ^ 1, kk + 32);   // overlap next tile

    // ---- scores: two 16-key C tiles -------------------------------------
    v8f c0 = {}, c1 = {};
#pragma unroll
    for (int ch = 0; ch < 4; ++ch) {
      BF16x16 b0, b1;
      const __bf16* p0 = &Kt[buf][lcol][ch * 32 + lhalf * 8];
      const __bf16* p1 = &Kt[buf][16 + lcol][ch * 32 + lhalf * 8];
      b0.h[0] = *(const v8bf*)p0;  b0.h[1] = *(const v8bf*)(p0 + 16);
      b1.h[0] = *(const v8bf*)p1;  b1.h[1] = *(const v8bf*)(p1 + 16);
      c0 = __builtin_amdgcn_wmma_f32_16x16x32_bf16(false, qf[ch].v, false, b0.v,
                                                   (short)0, c0, false, false);
      c1 = __builtin_amdgcn_wmma_f32_16x16x32_bf16(false, qf[ch].v, false, b1.v,
                                                   (short)0, c1, false, false);
    }

    // ---- online softmax (per row r: reduce across 16-lane half) ---------
#pragma unroll
    for (int r = 0; r < 8; ++r) {
      float s0 = c0[r] * tsv[r];
      float s1 = c1[r] * tsv[r];
      float mx = fmaxf(s0, s1);
      mx = fmaxf(mx, __shfl_xor(mx, 1, 32));
      mx = fmaxf(mx, __shfl_xor(mx, 2, 32));
      mx = fmaxf(mx, __shfl_xor(mx, 4, 32));
      mx = fmaxf(mx, __shfl_xor(mx, 8, 32));
      float nm = fmaxf(mrow[r], mx);
      float sc = __expf(mrow[r] - nm);
      mrow[r] = nm;
      float p0 = __expf(s0 - nm);
      float p1 = __expf(s1 - nm);
      float rs = p0 + p1;
      rs += __shfl_xor(rs, 1, 32);
      rs += __shfl_xor(rs, 2, 32);
      rs += __shfl_xor(rs, 4, 32);
      rs += __shfl_xor(rs, 8, 32);
      lrow[r] = lrow[r] * sc + rs;
#pragma unroll
      for (int dt = 0; dt < 8; ++dt) acc[dt][r] *= sc;
      // C-layout -> LDS (row = r + 8*lhalf, col = key within tile)
      Pt[wave][lhalf * 8 + r][lcol]      = (__bf16)p0;
      Pt[wave][lhalf * 8 + r][16 + lcol] = (__bf16)p1;
    }
    asm volatile("s_wait_dscnt 0" ::: "memory");  // wave-local LDS RAW

    // ---- P as A-fragment (16x32), then PV over 8 d-tiles ----------------
    BF16x16 pa;
    {
      const __bf16* pp = &Pt[wave][lcol][lhalf * 8];
      pa.h[0] = *(const v8bf*)pp;
      pa.h[1] = *(const v8bf*)(pp + 16);
    }
#pragma unroll
    for (int dt = 0; dt < 8; ++dt) {
      BF16x16 vb;
      const __bf16* vp = &Vt[buf][dt * 16 + lcol][lhalf * 8];
      vb.h[0] = *(const v8bf*)vp;
      vb.h[1] = *(const v8bf*)(vp + 16);
      acc[dt] = __builtin_amdgcn_wmma_f32_16x16x32_bf16(false, pa.v, false, vb.v,
                                                        (short)0, acc[dt], false, false);
    }
  }

  // ---- epilogue: normalize and store f32 --------------------------------
#pragma unroll
  for (int dt = 0; dt < 8; ++dt) {
    int dcol = dt * 16 + lcol;
#pragma unroll
    for (int r = 0; r < 8; ++r) {
      long row = q0 + lhalf * 8 + r;
      out[row * DIM + dcol] = acc[dt][r] / lrow[r];
    }
  }
}

// ---------------------------------------------------------------------------
extern "C" void kernel_launch(void* const* d_in, const int* in_sizes, int n_in,
                              void* d_out, int out_size, void* d_ws, size_t ws_size,
                              hipStream_t stream) {
  const float* x  = (const float*)d_in[0];
  const float* tg = (const float*)d_in[1];
  const float* Wq = (const float*)d_in[2];  const float* bq = (const float*)d_in[3];
  const float* Wk = (const float*)d_in[4];  const float* bk = (const float*)d_in[5];
  const float* Wv = (const float*)d_in[6];  const float* bv = (const float*)d_in[7];
  const float* W1 = (const float*)d_in[8];  const float* b1 = (const float*)d_in[9];
  const float* W2 = (const float*)d_in[10]; const float* b2 = (const float*)d_in[11];
  const float* W3 = (const float*)d_in[12]; const float* b3 = (const float*)d_in[13];
  float* out = (float*)d_out;

  const size_t QBYTES = (size_t)BB * NN * DIM * sizeof(__bf16);  // 8 MiB each
  char* ws = (char*)d_ws;
  __bf16* Qb = (__bf16*)(ws);
  __bf16* Kb = (__bf16*)(ws + QBYTES);
  __bf16* Vb = (__bf16*)(ws + 2 * QBYTES);   // transposed [B][DIM][N]
  float*  tb = (float*)(ws + 3 * QBYTES);

  ts_kernel  <<<(BB * NN) / 256, 256, 0, stream>>>(tg, W1, b1, W2, b2, W3, b3, tb);
  proj_kernel<<<(BB * NN) / 64, 128, 0, stream>>>(x, Wq, bq, Wk, bk, Wv, bv, Qb, Kb, Vb);
  attn_kernel<<<BB * (NN / 64), 128, 0, stream>>>(Qb, Kb, Vb, tb, out);
}
